// QuantumClassifier_9251359555842
// MI455X (gfx1250) — matrix operand order repaired and checked by reading, not compiled, not run on hardware
//
#include <hip/hip_runtime.h>
#include <math.h>

typedef __attribute__((ext_vector_type(16))) _Float16 v16h;
typedef __attribute__((ext_vector_type(8)))  float    v8f;

#define PI_F 3.14159265358979323846f

struct Cplx { float re, im; };

// n = a*x + b*y (complex fused form)
__device__ __forceinline__ Cplx cmad(Cplx a, Cplx x, Cplx b, Cplx y) {
  return Cplx{ a.re*x.re - a.im*x.im + b.re*y.re - b.im*y.im,
               a.re*x.im + a.im*x.re + b.re*y.im + b.im*y.re };
}

__device__ __forceinline__ float wave_sum(float v) {
#pragma unroll
  for (int off = 16; off > 0; off >>= 1) v += __shfl_xor(v, off, 32);
  return v;
}

// ---------------------------------------------------------------------------
// Kernel 1: x = x1 @ pre_w.T + pre_b  via v_wmma_f32_16x16x32_f16
// One wave computes a 16-row tile; N padded 4 -> 16 with zero columns.
// ---------------------------------------------------------------------------
__global__ void __launch_bounds__(128) pre_gemm_wmma(
    const float* __restrict__ x1,     // [B,96]
    const float* __restrict__ pre_w,  // [4,96]  (pre_w.T column n == pre_w row n)
    const float* __restrict__ pre_b,  // [4]
    float* __restrict__ xang,         // [B,4]
    int B)
{
  const int lane = threadIdx.x & 31;
  const int wave = blockIdx.x * (blockDim.x >> 5) + (threadIdx.x >> 5);
  const int row0 = wave * 16;
  if (row0 >= B) return;                 // wave-uniform; EXEC stays all-ones

  const int col    = lane & 15;
  const int a_koff = (lane < 16) ? 0 : 8;    // A 16x32 f16 layout (ISA 7.12.2)
  const int b_koff = (lane < 16) ? 0 : 8;    // B 32x16 f16 layout (same k order as A)

  v8f c = {};
#pragma unroll
  for (int kc = 0; kc < 96; kc += 32) {
    v16h a, b;
    const float* ap = x1 + (size_t)(row0 + col) * 96 + kc;
#pragma unroll
    for (int j = 0; j < 8; ++j) {
      a[j]     = (_Float16)ap[a_koff + j];
      a[8 + j] = (_Float16)ap[a_koff + 16 + j];
    }
#pragma unroll
    for (int j = 0; j < 16; ++j) b[j] = (_Float16)0.f;
    if (col < 4) {
      const float* bp = pre_w + col * 96 + kc + b_koff;
#pragma unroll
      for (int j = 0; j < 16; ++j) b[j] = (_Float16)bp[(j < 8) ? j : j + 8];
    }
    c = __builtin_amdgcn_wmma_f32_16x16x32_f16(false, a, false, b,
                                               (short)0, c, false, false);
  }
  // C/D layout: VGPR v -> (M=v, N=lane) for lanes 0-15, (M=v+8, N=lane-16) else
  if (col < 4) {
    const float bias  = pre_b[col];
    const int   mbase = (lane < 16) ? 0 : 8;
#pragma unroll
    for (int v = 0; v < 8; ++v)
      xang[(size_t)(row0 + mbase + v) * 4 + col] = c[v] + bias;
  }
}

// ---------------------------------------------------------------------------
// 8-qubit simulator: one sample per wave32.
// Amplitude index i in [0,256): lane = i>>3, slot = i&7.
// Wire w has bit position b = 7-w: b>=3 -> lane bit (b-3), else slot bit.
// Controlled-U on target wire tw with optional control wire cw (-1 = none).
// ---------------------------------------------------------------------------
__device__ __forceinline__ void apply_gate(
    float sre[8], float sim[8], int lane, int tw,
    Cplx u00, Cplx u01, Cplx u10, Cplx u11, int cw)
{
  const int tb = 7 - tw;
  if (tb >= 3) {                                 // cross-lane pair
    const int  lmask = 1 << (tb - 3);
    const bool hi    = (lane & lmask) != 0;
    const Cplx um = hi ? u11 : u00;              // coeff for my amplitude
    const Cplx up = hi ? u10 : u01;              // coeff for partner amplitude
#pragma unroll
    for (int s = 0; s < 8; ++s) {
      float prr = __shfl_xor(sre[s], lmask, 32); // all lanes execute the permute
      float pii = __shfl_xor(sim[s], lmask, 32);
      bool act = true;
      if (cw >= 0) {
        const int cb = 7 - cw;
        const int bit = (cb >= 3) ? ((lane >> (cb - 3)) & 1) : ((s >> cb) & 1);
        act = (bit != 0);
      }
      Cplx nv = cmad(um, Cplx{sre[s], sim[s]}, up, Cplx{prr, pii});
      if (act) { sre[s] = nv.re; sim[s] = nv.im; }
    }
  } else {                                       // intra-lane pair
    const int smask = 1 << tb;
#pragma unroll
    for (int s0 = 0; s0 < 8; ++s0) {
      if (s0 & smask) continue;
      const int s1 = s0 | smask;
      bool act = true;
      if (cw >= 0) {
        const int cb = 7 - cw;
        const int bit = (cb >= 3) ? ((lane >> (cb - 3)) & 1) : ((s0 >> cb) & 1);
        act = (bit != 0);
      }
      Cplx a0{sre[s0], sim[s0]}, a1{sre[s1], sim[s1]};
      Cplx n0 = cmad(u00, a0, u01, a1);
      Cplx n1 = cmad(u10, a0, u11, a1);
      if (act) { sre[s0]=n0.re; sim[s0]=n0.im; sre[s1]=n1.re; sim[s1]=n1.im; }
    }
  }
}

__device__ __forceinline__ void init_qkv8(
    float sre[8], float sim[8], int lane,
    const float* __restrict__ rot, const float* __restrict__ crx, int off)
{
  const int pa[8] = {0,1,2,3,0,1,2,3};
  const int pb[8] = {1,2,3,0,3,0,1,2};
#pragma unroll
  for (int i = 0; i < 8; ++i) {
    const int wa = off + pa[i], wb = off + pb[i];
    // Rot(phi,theta,omega) = RZ(om) RY(th) RZ(phi)
    const float phi = rot[3*i], th = rot[3*i+1], om = rot[3*i+2];
    const float ch = cosf(0.5f*th), sh = sinf(0.5f*th);
    const float ap = 0.5f*(phi+om), am = 0.5f*(phi-om);
    const float cap = cosf(ap), sap = sinf(ap), cam = cosf(am), sam = sinf(am);
    apply_gate(sre, sim, lane, wa,
               Cplx{ cap*ch, -sap*ch}, Cplx{-cam*sh, -sam*sh},
               Cplx{ cam*sh, -sam*sh}, Cplx{ cap*ch,  sap*ch}, -1);
    // CRX(t): ctrl wa, tgt wb
    const float t = crx[i];
    const float c = cosf(0.5f*t), s = sinf(0.5f*t);
    apply_gate(sre, sim, lane, wb,
               Cplx{c,0.f}, Cplx{0.f,-s}, Cplx{0.f,-s}, Cplx{c,0.f}, wa);
    // CNOT: ctrl wa, tgt wb
    apply_gate(sre, sim, lane, wb,
               Cplx{0.f,0.f}, Cplx{1.f,0.f}, Cplx{1.f,0.f}, Cplx{0.f,0.f}, wa);
  }
}

// 4-qubit scalar simulator (replicated across lanes; inputs wave-uniform).
__device__ __forceinline__ void apply4(
    float pr[16], float pq[16], int tw,
    Cplx u00, Cplx u01, Cplx u10, Cplx u11, int cw)
{
  const int tm = 1 << (3 - tw);
#pragma unroll
  for (int i0 = 0; i0 < 16; ++i0) {
    if (i0 & tm) continue;
    const int i1 = i0 | tm;
    if (cw >= 0 && !((i0 >> (3 - cw)) & 1)) continue;
    Cplx a0{pr[i0], pq[i0]}, a1{pr[i1], pq[i1]};
    Cplx n0 = cmad(u00, a0, u01, a1);
    Cplx n1 = cmad(u10, a0, u11, a1);
    pr[i0]=n0.re; pq[i0]=n0.im; pr[i1]=n1.re; pq[i1]=n1.im;
  }
}

__global__ void __launch_bounds__(256) qsim_kernel(
    const float* __restrict__ xang,   // [B,4]
    const float* __restrict__ q_rot, const float* __restrict__ q_crx,
    const float* __restrict__ k_rot, const float* __restrict__ k_crx,
    const float* __restrict__ v_rot, const float* __restrict__ v_crx,
    const float* __restrict__ ln_w,  const float* __restrict__ ln_b,
    const float* __restrict__ head_w, const float* __restrict__ head_b,
    float* __restrict__ out, int B)
{
  const int lane = threadIdx.x & 31;
  const int samp = blockIdx.x * (blockDim.x >> 5) + (threadIdx.x >> 5);
  if (samp >= B) return;

  float xs[4];
#pragma unroll
  for (int w = 0; w < 4; ++w) xs[w] = xang[(size_t)samp * 4 + w];

  // ---- 8-qubit score circuit ----
  float sre[8], sim[8];
#pragma unroll
  for (int s = 0; s < 8; ++s) { sre[s] = 0.f; sim[s] = 0.f; }
  if (lane == 0) sre[0] = 1.f;                     // |00000000>

#pragma unroll
  for (int w = 0; w < 4; ++w) {                    // AngleEmbedding RX(xq)
    const float c = cosf(0.5f*xs[w]), s = sinf(0.5f*xs[w]);
    apply_gate(sre, sim, lane, w, Cplx{c,0.f}, Cplx{0.f,-s}, Cplx{0.f,-s}, Cplx{c,0.f}, -1);
  }
  init_qkv8(sre, sim, lane, q_rot, q_crx, 0);
#pragma unroll
  for (int w = 0; w < 4; ++w) {                    // AngleEmbedding RX(xk)
    const float c = cosf(0.5f*xs[w]), s = sinf(0.5f*xs[w]);
    apply_gate(sre, sim, lane, w, Cplx{c,0.f}, Cplx{0.f,-s}, Cplx{0.f,-s}, Cplx{c,0.f}, -1);
  }
  init_qkv8(sre, sim, lane, k_rot, k_crx, 4);
#pragma unroll
  for (int w = 0; w < 4; ++w)                      // CNOT(w, w+4)
    apply_gate(sre, sim, lane, w + 4,
               Cplx{0.f,0.f}, Cplx{1.f,0.f}, Cplx{1.f,0.f}, Cplx{0.f,0.f}, w);

  float score[4];
#pragma unroll
  for (int w = 0; w < 4; ++w) {
    const int lm = 1 << (4 - w);                   // wires 0..3 are lane bits
    float az = 0.f, ax = 0.f;
#pragma unroll
    for (int s = 0; s < 8; ++s) {
      const float p = sre[s]*sre[s] + sim[s]*sim[s];
      az += ((lane >> (4 - w)) & 1) ? -p : p;      // <Z>
      const float prr = __shfl_xor(sre[s], lm, 32);
      const float pii = __shfl_xor(sim[s], lm, 32);
      ax += sre[s]*prr + sim[s]*pii;               // <X> = Re sum conj(psi)*flip
    }
    az = wave_sum(az);
    ax = wave_sum(ax);
    score[w] = sqrtf(az*az + ax*ax);               // wave-uniform now
  }

  // ---- 4-qubit value circuit (replicated per lane; all inputs uniform) ----
  float pr[16], pq[16];
#pragma unroll
  for (int i = 0; i < 16; ++i) { pr[i] = 0.f; pq[i] = 0.f; }
  pr[0] = 1.f;

#pragma unroll
  for (int w = 0; w < 4; ++w) {
    const float c = cosf(0.5f*xs[w]), s = sinf(0.5f*xs[w]);
    apply4(pr, pq, w, Cplx{c,0.f}, Cplx{0.f,-s}, Cplx{0.f,-s}, Cplx{c,0.f}, -1);
  }
  {
    const int pa[8] = {0,1,2,3,0,1,2,3};
    const int pb[8] = {1,2,3,0,3,0,1,2};
#pragma unroll
    for (int i = 0; i < 8; ++i) {
      const float phi = v_rot[3*i], th = v_rot[3*i+1], om = v_rot[3*i+2];
      const float ch = cosf(0.5f*th), sh = sinf(0.5f*th);
      const float ap = 0.5f*(phi+om), am = 0.5f*(phi-om);
      const float cap = cosf(ap), sap = sinf(ap), cam = cosf(am), sam = sinf(am);
      apply4(pr, pq, pa[i],
             Cplx{ cap*ch, -sap*ch}, Cplx{-cam*sh, -sam*sh},
             Cplx{ cam*sh, -sam*sh}, Cplx{ cap*ch,  sap*ch}, -1);
      const float t = v_crx[i];
      const float c = cosf(0.5f*t), s = sinf(0.5f*t);
      apply4(pr, pq, pb[i], Cplx{c,0.f}, Cplx{0.f,-s}, Cplx{0.f,-s}, Cplx{c,0.f}, pa[i]);
      apply4(pr, pq, pb[i], Cplx{0.f,0.f}, Cplx{1.f,0.f}, Cplx{1.f,0.f}, Cplx{0.f,0.f}, pa[i]);
    }
  }
#pragma unroll
  for (int w = 0; w < 4; ++w) {                    // RZ(tanh(score)*pi)
    const float t = tanhf(score[w]) * PI_F;
    const float c = cosf(0.5f*t), s = sinf(0.5f*t);
    apply4(pr, pq, w, Cplx{c,-s}, Cplx{0.f,0.f}, Cplx{0.f,0.f}, Cplx{c,s}, -1);
  }
#pragma unroll
  for (int a = 0; a < 3; ++a)                      // CNOT chain (0,1)(1,2)(2,3)
    apply4(pr, pq, a + 1, Cplx{0.f,0.f}, Cplx{1.f,0.f}, Cplx{1.f,0.f}, Cplx{0.f,0.f}, a);

  float outs[8];
#pragma unroll
  for (int w = 0; w < 4; ++w) {
    const int m = 1 << (3 - w);
    float az = 0.f, ax = 0.f;
#pragma unroll
    for (int i = 0; i < 16; ++i) {
      const float p = pr[i]*pr[i] + pq[i]*pq[i];
      az += (i & m) ? -p : p;
      ax += pr[i]*pr[i ^ m] + pq[i]*pq[i ^ m];
    }
    outs[w] = az; outs[4 + w] = ax;
  }

  // ---- LayerNorm(8) + exact GELU + head [8]->[2] ----
  float mu = 0.f;
#pragma unroll
  for (int k = 0; k < 8; ++k) mu += outs[k];
  mu *= 0.125f;
  float var = 0.f;
#pragma unroll
  for (int k = 0; k < 8; ++k) { const float d = outs[k] - mu; var += d*d; }
  var *= 0.125f;
  const float inv = 1.f / sqrtf(var + 1e-5f);
  float o0 = head_b[0], o1 = head_b[1];
#pragma unroll
  for (int k = 0; k < 8; ++k) {
    const float y = (outs[k] - mu) * inv * ln_w[k] + ln_b[k];
    const float g = 0.5f * y * (1.f + erff(y * 0.70710678118654752f));
    o0 += g * head_w[k];
    o1 += g * head_w[8 + k];
  }
  if (lane == 0) {
    out[(size_t)samp * 2 + 0] = o0;
    out[(size_t)samp * 2 + 1] = o1;
  }
}

// ---------------------------------------------------------------------------
extern "C" void kernel_launch(void* const* d_in, const int* in_sizes, int n_in,
                              void* d_out, int out_size, void* d_ws, size_t ws_size,
                              hipStream_t stream) {
  const float* x1     = (const float*)d_in[0];
  const float* pre_w  = (const float*)d_in[1];
  const float* pre_b  = (const float*)d_in[2];
  const float* q_rot  = (const float*)d_in[3];
  const float* k_rot  = (const float*)d_in[4];
  const float* v_rot  = (const float*)d_in[5];
  const float* q_crx  = (const float*)d_in[6];
  const float* k_crx  = (const float*)d_in[7];
  const float* v_crx  = (const float*)d_in[8];
  const float* ln_w   = (const float*)d_in[9];
  const float* ln_b   = (const float*)d_in[10];
  const float* head_w = (const float*)d_in[11];
  const float* head_b = (const float*)d_in[12];
  float* outp = (float*)d_out;

  const int B = in_sizes[0] / 96;          // 16384
  float* xang = (float*)d_ws;              // [B,4] scratch angles

  // Kernel 1: WMMA pre-projection. 1 wave per 16 rows; 4 waves per 128-thr block.
  const int waves1  = (B + 15) / 16;
  const int blocks1 = (waves1 + 3) / 4;
  pre_gemm_wmma<<<blocks1, 128, 0, stream>>>(x1, pre_w, pre_b, xang, B);

  // Kernel 2: 1 sample per wave; 8 waves per 256-thr block.
  const int blocks2 = (B + 7) / 8;
  qsim_kernel<<<blocks2, 256, 0, stream>>>(xang,
      q_rot, q_crx, k_rot, k_crx, v_rot, v_crx,
      ln_w, ln_b, head_w, head_b, outp, B);
}
